// ChebConv_17841294148274
// MI455X (gfx1250) — compile-verified, matrix operand order checked by reading, not run on hardware
//
#include <hip/hip_runtime.h>
#include <hip/hip_bf16.h>
#include <hip/hip_fp16.h>

// ---------------------------------------------------------------------------
// ChebConv: xw = Xr(49152x64) @ W(64x192); then COO gather-scale-scatter-add
// into out(2048x1536) initialized with broadcast bias.
// MI455X / gfx1250: wave32, WMMA f32 16x16x4 for the GEMM, f32 L2 atomics for
// the scatter. xw lives in d_ws (37.75 MB, L2-resident on a 192MB-L2 part).
// ---------------------------------------------------------------------------

typedef __attribute__((ext_vector_type(2)))  float    v2f;
typedef __attribute__((ext_vector_type(8)))  float    v8f;
typedef __attribute__((ext_vector_type(16))) _Float16 v16h;

#define M_TOT    49152          // (B*C_IN*T*N)/64 rows of Xr
#define K_DIM    64
#define N_DIM    192            // KS*C_OUT
#define N_TILES  (N_DIM / 16)   // 12
#define M_TILES  (M_TOT / 16)   // 3072
#define TILES    (M_TILES * N_TILES)  // 36864 wave-tiles
#define XW_COLS  1536           // reshape(6144, 1536) of the same buffer
#define NNZ_CNT  98304
#define OUT_ELEMS (2048 * 1536) // == out_size

// ---------------------------------------------------------------------------
// Kernel 1: out[i] = bias[i % 64]  (accumulation target for the scatter)
// ---------------------------------------------------------------------------
__global__ __launch_bounds__(256)
void init_bias_kernel(const float* __restrict__ bias, float* __restrict__ out) {
    int i = blockIdx.x * 256 + threadIdx.x;
    if (i < OUT_ELEMS) out[i] = bias[i & 63];
}

// ---------------------------------------------------------------------------
// Kernel 2: xw(49152x192) = Xr(49152x64) @ W(64x192), one wave per 16x16 tile.
// ---------------------------------------------------------------------------
__global__ __launch_bounds__(256)
void gemm_xw_wmma(const float* __restrict__ x,   // flat (B,C_IN,T,N) == Xr rows
                  const float* __restrict__ w,   // flat (KS,C_IN,C_OUT) == 64x192
                  float* __restrict__ xw) {
    const int wave = blockIdx.x * (256 >> 5) + (threadIdx.x >> 5);
    if (wave >= TILES) return;
    const int lane = threadIdx.x & 31;
    const int half = lane >> 4;    // 0: lanes 0-15, 1: lanes 16-31
    const int lid  = lane & 15;
    const int m0 = (wave / N_TILES) * 16;
    const int n0 = (wave % N_TILES) * 16;

    v8f c = {};

#if __has_builtin(__builtin_amdgcn_wmma_f32_16x16x4_f32)
    // f32 A 16x4 layout: lanes 0-15 hold M=lid with K={k0,k0+1},
    //                    lanes 16-31 hold M=lid with K={k0+2,k0+3}.
    // B 4x16 mirrored:   lanes 0-15 hold N=lid with K={k0,k0+1}, etc.
    const float* arow = x + (size_t)(m0 + lid) * K_DIM + 2 * half;  // 8B aligned
    const float* bcol = w + (size_t)(2 * half) * N_DIM + (n0 + lid);
#pragma unroll
    for (int k0 = 0; k0 < K_DIM; k0 += 4) {
        v2f a = *reinterpret_cast<const v2f*>(arow + k0);           // K, K+1
        v2f b;
        b.x = bcol[(size_t)(k0 + 0) * N_DIM];                        // W[K  ][n]
        b.y = bcol[(size_t)(k0 + 1) * N_DIM];                        // W[K+1][n]
        c = __builtin_amdgcn_wmma_f32_16x16x4_f32(
                false, a, false, b, (short)0, c, false, false);
    }
#else
    // Fallback: f16 inputs, f32 accumulate (codegen-confirmed builtin).
    // 16-bit A 16x32 layout: VGPR j<4 -> K = k0 + 2j + 8*half,
    //                        VGPR j>=4 -> K = k0 + 16 + 2(j-4) + 8*half.
    const float* arow = x + (size_t)(m0 + lid) * K_DIM;
#pragma unroll
    for (int k0 = 0; k0 < K_DIM; k0 += 32) {
        v16h a, b;
#pragma unroll
        for (int j = 0; j < 8; ++j) {
            int kk = k0 + ((j < 4) ? (2 * j + 8 * half)
                                   : (16 + 2 * (j - 4) + 8 * half));
            a[2 * j]     = (_Float16)arow[kk];
            a[2 * j + 1] = (_Float16)arow[kk + 1];
            b[2 * j]     = (_Float16)w[(size_t)kk * N_DIM + n0 + lid];
            b[2 * j + 1] = (_Float16)w[(size_t)(kk + 1) * N_DIM + n0 + lid];
        }
        c = __builtin_amdgcn_wmma_f32_16x16x32_f16(
                false, a, false, b, (short)0, c, false, false);
    }
#endif

    // 16x16 f32 D layout: lanes 0-15 -> N=lid, M=v; lanes 16-31 -> N=lid, M=8+v.
    float* drow = xw + (size_t)(m0 + 8 * half) * N_DIM + (n0 + lid);
#pragma unroll
    for (int v = 0; v < 8; ++v)
        drow[(size_t)v * N_DIM] = c[v];
}

// ---------------------------------------------------------------------------
// Kernel 3: per-nonzero gather/scale/scatter-add. One 256-thread block per
// nonzero; 6 f32 atomics per lane. xw is L2-resident (37.7MB << 192MB L2).
// ---------------------------------------------------------------------------
__global__ __launch_bounds__(256)
void spmm_scatter(const float* __restrict__ xw,
                  const float* __restrict__ vals,
                  const int*   __restrict__ rows,
                  const int*   __restrict__ cols,
                  float* __restrict__ out) {
    const int e   = blockIdx.x;
    const float v = vals[e];
    const float* __restrict__ src = xw  + (size_t)cols[e] * XW_COLS;
    float* __restrict__       dst = out + (size_t)rows[e] * XW_COLS;
#pragma unroll
    for (int it = 0; it < XW_COLS / 256; ++it) {
        int j = threadIdx.x + it * 256;
        unsafeAtomicAdd(&dst[j], v * src[j]);   // -> global_atomic_add_f32
    }
}

// ---------------------------------------------------------------------------
extern "C" void kernel_launch(void* const* d_in, const int* in_sizes, int n_in,
                              void* d_out, int out_size, void* d_ws, size_t ws_size,
                              hipStream_t stream) {
    const float* x    = (const float*)d_in[0];  // (2,64,12,2048) f32
    const float* w    = (const float*)d_in[1];  // (3,64,64) f32 == 64x192
    const float* bias = (const float*)d_in[2];  // (64,) f32
    const float* vals = (const float*)d_in[3];  // (98304,) f32
    const int*   rows = (const int*)d_in[4];    // (98304,) i32 in [0,2048)
    const int*   cols = (const int*)d_in[5];    // (98304,) i32 in [0,6144)
    float* out = (float*)d_out;                 // 3,145,728 f32
    float* xw  = (float*)d_ws;                  // needs 37,748,736 bytes

    hipLaunchKernelGGL(init_bias_kernel, dim3((OUT_ELEMS + 255) / 256), dim3(256),
                       0, stream, bias, out);
    hipLaunchKernelGGL(gemm_xw_wmma, dim3((TILES + 7) / 8), dim3(256),
                       0, stream, x, w, xw);
    hipLaunchKernelGGL(spmm_scatter, dim3(NNZ_CNT), dim3(256),
                       0, stream, xw, vals, rows, cols, out);
}
